// UVInstantNGP_19378892440259
// MI455X (gfx1250) — matrix operand order repaired and checked by reading, not compiled, not run on hardware
//
#include <hip/hip_runtime.h>
#include <hip/hip_fp16.h>
#include <stdint.h>

typedef __attribute__((ext_vector_type(16))) _Float16 v16h;
typedef __attribute__((ext_vector_type(8)))  _Float16 v8h;
typedef __attribute__((ext_vector_type(8)))  float    v8f;

#define HH 1024
#define WW 1024
#define LOG2T 19
#define TSZ (1u << LOG2T)

static __device__ __forceinline__ uint32_t pack2h(float a, float b) {
  __half2 p = __floats2half2_rn(a, b);    // v_cvt_pk_f16_f32
  union { __half2 h; uint32_t u; } c; c.h = p;
  return c.u;
}

union AFrag  { v16h v; uint32_t u[8]; uint4 q[2]; };
union CFragH { v8h  v; uint32_t u[4]; };

// single-instruction packed ReLU: v_pk_max_num_f16 dst, src, 0
static __device__ __forceinline__ void relu_pk4(CFragH& c) {
#pragma unroll
  for (int i = 0; i < 4; i++)
    asm("v_pk_max_num_f16 %0, %1, 0" : "=v"(c.u[i]) : "v"(c.u[i]));
}

// floor(16 * (2048/16)^(l/15)) for l = 0..15, split by parity
__constant__ const float RES_EVEN[8] = {16.f, 30.f, 58.f, 111.f, 212.f, 406.f, 776.f, 1482.f};
__constant__ const float RES_ODD[8]  = {22.f, 42.f, 80.f, 154.f, 294.f, 561.f, 1072.f, 2048.f};

__launch_bounds__(256)
__global__ void ngp_fused_kernel(const float* __restrict__ tex,
                                 const float* __restrict__ htab,
                                 const float* __restrict__ w1, const float* __restrict__ b1,
                                 const float* __restrict__ w2, const float* __restrict__ b2,
                                 const float* __restrict__ w3, const float* __restrict__ b3,
                                 float* __restrict__ out) {
  // 14 pre-swizzled f16 B-fragments (K32 x N16), each: 32 lanes * 8 dwords
  __shared__ uint32_t sW[14 * 256];
  // per-wave staging (8 waves / 256-thread block), no cross-wave sharing
  __shared__ uint32_t sEnc[8][256];                    // enc[level][pixel] packed f16x2
  __shared__ __align__(16) _Float16 sH1[8][16 * 64];   // h1[m][feat]
  __shared__ __align__(16) _Float16 sH2[8][16 * 64];   // h2[m][feat]

  const int tid = threadIdx.x;

  // ---- stage all weights once as WMMA B-fragments (f16) ----
  // frag 0..3  : w1  (ntile = f,          kchunk = 0)
  // frag 4..11 : w2  (ntile = (f-4)/2,    kchunk = (f-4)&1)
  // frag 12..13: w3  (N padded 3 -> 16,   kchunk = f-12)
  for (int e = tid; e < 14 * 256; e += 256) {
    int f = e >> 8, lane = (e >> 3) & 31, j = e & 7;
    int nn = lane & 15;
    int kb = ((lane >> 4) << 3) + 2 * j + ((j >> 2) << 3);   // table-order k pair within K32 chunk
    float v0, v1;
    if (f < 4) {
      int n = f * 16 + nn;
      v0 = w1[kb * 64 + n];
      v1 = w1[(kb + 1) * 64 + n];
    } else if (f < 12) {
      int g = f - 4; int nt = g >> 1; int kc = g & 1;
      int n = nt * 16 + nn; int k = kc * 32 + kb;
      v0 = w2[k * 64 + n];
      v1 = w2[(k + 1) * 64 + n];
    } else {
      int kc = f - 12; int k = kc * 32 + kb;
      v0 = (nn < 3) ? w3[k * 3 + nn] : 0.0f;
      v1 = (nn < 3) ? w3[(k + 1) * 3 + nn] : 0.0f;
    }
    sW[e] = pack2h(v0, v1);
  }
  __syncthreads();

  const int lane = tid & 31;
  const int wib  = tid >> 5;
  const int mrow = lane & 15;
  const int hi   = lane >> 4;        // 0: lanes 0-15, 1: lanes 16-31
  const int mb   = hi * 8;           // C-layout row base for this lane

  // biases (f16 for hidden layers, f32 for output layer)
  _Float16 bh1[4], bh2[4];
#pragma unroll
  for (int nt = 0; nt < 4; nt++) {
    bh1[nt] = (_Float16)b1[nt * 16 + mrow];
    bh2[nt] = (_Float16)b2[nt * 16 + mrow];
  }
  const float bv3 = (mrow < 3) ? b3[mrow] : 0.0f;

  uint32_t*  enc = sEnc[wib];
  _Float16*  h1  = sH1[wib];
  _Float16*  h2  = sH2[wib];
  const uint4*  wfrag = (const uint4*)&sW[lane * 8];  // + f*64 uint4s per fragment
  const float2* ht2   = (const float2*)htab;          // 2^23 float2 entries total

  const int wavesTotal = gridDim.x * 8;
  const int waveId     = blockIdx.x * 8 + wib;

  for (int t = waveId; t < (HH * WW / 16); t += wavesTotal) {
    const int p0 = t << 4;
    const int x0 = p0 & (WW - 1);
    const int y  = p0 >> 10;
    const float vcoord = (float)y * (1.0f / HH);
    const float ucoord = (float)(x0 + mrow) * (1.0f / WW);

    // ---- hash-grid encode: lane does (pixel=mrow, level=2j+hi) ----
#pragma unroll
    for (int j = 0; j < 8; j++) {
      const int level = 2 * j + hi;
      const float resf = hi ? RES_ODD[j] : RES_EVEN[j];   // j const after unroll
      const float sx = ucoord * resf, sy = vcoord * resf;
      const float fx0 = floorf(sx), fy0 = floorf(sy);
      const float fx = sx - fx0, fy = sy - fy0;
      const uint32_t ix = (uint32_t)fx0, iy = (uint32_t)fy0;
      const uint32_t lb32 = (uint32_t)level << LOG2T;     // float2-element base
      const uint32_t PR = 2654435761u;
      const uint32_t i00 = lb32 + (( ix       ^ ( iy       * PR)) & (TSZ - 1u));
      const uint32_t i01 = lb32 + (( ix       ^ ((iy + 1u) * PR)) & (TSZ - 1u));
      const uint32_t i10 = lb32 + (((ix + 1u) ^ ( iy       * PR)) & (TSZ - 1u));
      const uint32_t i11 = lb32 + (((ix + 1u) ^ ((iy + 1u) * PR)) & (TSZ - 1u));
      const float2 f00 = ht2[i00];
      const float2 f01 = ht2[i01];
      const float2 f10 = ht2[i10];
      const float2 f11 = ht2[i11];
      // two-axis lerp (packed f32 FMAs)
      const float ax = f00.x + fx * (f10.x - f00.x);
      const float bx = f01.x + fx * (f11.x - f01.x);
      const float ay = f00.y + fx * (f10.y - f00.y);
      const float by = f01.y + fx * (f11.y - f01.y);
      const float e0 = ax + fy * (bx - ax);
      const float e1 = ay + fy * (by - ay);
      enc[level * 16 + mrow] = pack2h(e0, e1);  // bank-conflict-free
    }

    // ---- A1 fragment (16x32 f16): lane<16 -> levels {0..3, 8..11}, lane>=16 -> {4..7, 12..15}
    AFrag a1;
    const int lb = hi * 4;
#pragma unroll
    for (int r = 0; r < 4; r++) {
      a1.u[r]     = enc[(lb + r)     * 16 + mrow];
      a1.u[4 + r] = enc[(lb + 8 + r) * 16 + mrow];
    }

    // ---- layer 1: [16x32] @ [32x64], 4 WMMAs, f16 accumulate ----
#pragma unroll
    for (int nt = 0; nt < 4; nt++) {
      AFrag bw;
      bw.q[0] = wfrag[nt * 64];
      bw.q[1] = wfrag[nt * 64 + 1];
      CFragH c;
#pragma unroll
      for (int h = 0; h < 8; h++) c.v[h] = bh1[nt];
      c.v = __builtin_amdgcn_wmma_f16_16x16x32_f16(false, a1.v, false, bw.v,
                                                   (short)0, c.v, false, false);
      relu_pk4(c);                               // 4x v_pk_max_num_f16
      const int n = nt * 16 + mrow;
#pragma unroll
      for (int h = 0; h < 8; h++) h1[(mb + h) * 64 + n] = c.v[h];
    }

    // ---- A2 fragments (16x64 f16 as two K32 chunks) ----
    AFrag a2a, a2b;
    {
      const _Float16* hp = h1 + mrow * 64;
      const int ko = hi * 8;
      a2a.q[0] = *(const uint4*)(hp + ko);
      a2a.q[1] = *(const uint4*)(hp + 16 + ko);
      a2b.q[0] = *(const uint4*)(hp + 32 + ko);
      a2b.q[1] = *(const uint4*)(hp + 48 + ko);
    }

    // ---- layer 2: [16x64] @ [64x64], 8 WMMAs, f16 accumulate ----
#pragma unroll
    for (int nt = 0; nt < 4; nt++) {
      AFrag bw0, bw1;
      bw0.q[0] = wfrag[(4 + nt * 2) * 64];
      bw0.q[1] = wfrag[(4 + nt * 2) * 64 + 1];
      bw1.q[0] = wfrag[(5 + nt * 2) * 64];
      bw1.q[1] = wfrag[(5 + nt * 2) * 64 + 1];
      CFragH c;
#pragma unroll
      for (int h = 0; h < 8; h++) c.v[h] = bh2[nt];
      c.v = __builtin_amdgcn_wmma_f16_16x16x32_f16(false, a2a.v, false, bw0.v,
                                                   (short)0, c.v, false, false);
      c.v = __builtin_amdgcn_wmma_f16_16x16x32_f16(false, a2b.v, false, bw1.v,
                                                   (short)0, c.v, false, false);
      relu_pk4(c);
      const int n = nt * 16 + mrow;
#pragma unroll
      for (int h = 0; h < 8; h++) h2[(mb + h) * 64 + n] = c.v[h];
    }

    // ---- A3 fragments ----
    AFrag a3a, a3b;
    {
      const _Float16* hp = h2 + mrow * 64;
      const int ko = hi * 8;
      a3a.q[0] = *(const uint4*)(hp + ko);
      a3a.q[1] = *(const uint4*)(hp + 16 + ko);
      a3b.q[0] = *(const uint4*)(hp + 32 + ko);
      a3b.q[1] = *(const uint4*)(hp + 48 + ko);
    }

    // ---- layer 3: [16x64] @ [64x16(pad of 3)], 2 WMMAs, f32 accumulate ----
    AFrag bw3a, bw3b;
    bw3a.q[0] = wfrag[12 * 64];
    bw3a.q[1] = wfrag[12 * 64 + 1];
    bw3b.q[0] = wfrag[13 * 64];
    bw3b.q[1] = wfrag[13 * 64 + 1];
    v8f c3;
#pragma unroll
    for (int r = 0; r < 8; r++) c3[r] = bv3;
    c3 = __builtin_amdgcn_wmma_f32_16x16x32_f16(false, a3a.v, false, bw3a.v,
                                                (short)0, c3, false, false);
    c3 = __builtin_amdgcn_wmma_f32_16x16x32_f16(false, a3b.v, false, bw3b.v,
                                                (short)0, c3, false, false);

    // ---- epilogue: residual + sigmoid(base), clip, coalesced b128 I/O ----
    if (mrow < 3) {
      const size_t base = (size_t)mrow * (HH * WW) + (size_t)y * WW + (size_t)(x0 + mb);
      const float4 t0 = *(const float4*)(tex + base);
      const float4 t1 = *(const float4*)(tex + base + 4);
      float o[8];
#pragma unroll
      for (int r = 0; r < 8; r++) {
        const float tv = (r < 4) ? ((const float*)&t0)[r] : ((const float*)&t1)[r - 4];
        const float sg = 1.0f / (1.0f + expf(-tv));
        const float rs = 0.25f * tanhf(c3[r]);
        const float ov = sg + rs;
        o[r] = fminf(fmaxf(ov, 0.0f), 1.0f);
      }
      *(float4*)(out + base)     = make_float4(o[0], o[1], o[2], o[3]);
      *(float4*)(out + base + 4) = make_float4(o[4], o[5], o[6], o[7]);
    }
  }
}

extern "C" void kernel_launch(void* const* d_in, const int* in_sizes, int n_in,
                              void* d_out, int out_size, void* d_ws, size_t ws_size,
                              hipStream_t stream) {
  (void)in_sizes; (void)n_in; (void)out_size; (void)d_ws; (void)ws_size;
  const float* tex = (const float*)d_in[0];
  const float* ht  = (const float*)d_in[1];
  const float* w1  = (const float*)d_in[2];
  const float* b1  = (const float*)d_in[3];
  const float* w2  = (const float*)d_in[4];
  const float* b2  = (const float*)d_in[5];
  const float* w3  = (const float*)d_in[6];
  const float* b3  = (const float*)d_in[7];
  // 65536 pixel-tiles of 16; 512 blocks * 8 waves = 4096 waves -> 16 tiles/wave
  ngp_fused_kernel<<<512, 256, 0, stream>>>(tex, ht, w1, b1, w2, b2, w3, b3,
                                            (float*)d_out);
}
